// FusionMultiheadedAttention_44117904064540
// MI455X (gfx1250) — compile-verified, matrix operand order checked by reading, not run on hardware
//
#include <hip/hip_runtime.h>
#include <hip/hip_bf16.h>

typedef __bf16 bf16;
typedef __attribute__((ext_vector_type(16))) __bf16 bf16x16;
typedef __attribute__((ext_vector_type(8)))  __bf16 bf16x8;
typedef __attribute__((ext_vector_type(8)))  float  f32x8;

#define CAT16(lo, hi) __builtin_shufflevector((lo), (hi), 0,1,2,3,4,5,6,7,8,9,10,11,12,13,14,15)

static constexpr int  BB   = 8;
static constexpr int  SS   = 2048;
static constexpr int  DM   = 1024;
static constexpr int  HH   = 8;
static constexpr int  DK   = 128;
static constexpr int  DKV  = 144;
static constexpr int  NKV  = 1152;
static constexpr int  DKVP = 160;           // per-head kv dim padded to mult of 32
static constexpr long ROWS = (long)BB * SS; // 16384

__device__ __forceinline__ f32x8 wmma_bf16(bf16x16 a, bf16x16 b, f32x8 c) {
  return __builtin_amdgcn_wmma_f32_16x16x32_bf16(false, a, false, b, (short)0, c, false, false);
}

__device__ __forceinline__ void wait_async0() {
#if __has_builtin(__builtin_amdgcn_s_wait_asynccnt)
  __builtin_amdgcn_s_wait_asynccnt(0);
#else
  asm volatile("s_wait_asynccnt 0x0" ::: "memory");
#endif
}

// async copy 32B (two b128, INST_OFFSET applies to both LDS and global address)
__device__ __forceinline__ void async_cp32(unsigned ldsAddr, const void* gaddr) {
  asm volatile("global_load_async_to_lds_b128 %0, %1, off\n\t"
               "global_load_async_to_lds_b128 %0, %1, off offset:32"
               :: "v"(ldsAddr), "v"(gaddr) : "memory");
}

// ---------------- elementwise pack / convert ----------------

__global__ void cvt_f32_bf16(const float* __restrict__ src, bf16* __restrict__ dst, long n) {
  long i = (long)blockIdx.x * blockDim.x + threadIdx.x;
  long st = (long)gridDim.x * blockDim.x;
  for (; i < n; i += st) dst[i] = (bf16)src[i];
}

__global__ void pack_concat_bf16(const float* __restrict__ X, const float* __restrict__ Y,
                                 bf16* __restrict__ dst) {
  long n = ROWS * (long)NKV;
  long i = (long)blockIdx.x * blockDim.x + threadIdx.x;
  long st = (long)gridDim.x * blockDim.x;
  for (; i < n; i += st) {
    long r = i / NKV;
    int  c = (int)(i - r * NKV);
    float v = (c < DM) ? X[r * DM + c] : Y[r * 128 + (c - DM)];
    dst[i] = (bf16)v;
  }
}

// attnH [B,H,S,128] -> attnPacked [B*S, 1024]
__global__ void pack_attn(const bf16* __restrict__ src, bf16* __restrict__ dst) {
  long n = ROWS * (long)DM;
  long i = (long)blockIdx.x * blockDim.x + threadIdx.x;
  long st = (long)gridDim.x * blockDim.x;
  for (; i < n; i += st) {
    long r = i >> 10;
    int  c = (int)(i & (DM - 1));
    int  h = c >> 7, d = c & 127;
    long b = r >> 11;
    long s = r & (SS - 1);
    dst[i] = src[(((b * HH + h) * SS + s) << 7) + d];
  }
}

// ---------------- leaky_relu + log_softmax over 1152 ----------------
__global__ __launch_bounds__(256) void act_logsoftmax(const bf16* __restrict__ X,
                                                      bf16* __restrict__ Yflat,
                                                      bf16* __restrict__ Ypad,
                                                      int mode) {
  __shared__ float red[8];
  const long row = blockIdx.x;
  const bf16* x = X + row * NKV;
  const int t = threadIdx.x, wv = t >> 5, ln = t & 31;

  float mx = -3.0e38f;
  for (int i = t; i < NKV; i += 256) {
    float v = (float)x[i]; v = v >= 0.f ? v : 0.01f * v;
    mx = fmaxf(mx, v);
  }
  for (int off = 16; off; off >>= 1) mx = fmaxf(mx, __shfl_xor(mx, off, 32));
  if (ln == 0) red[wv] = mx;
  __syncthreads();
  mx = red[0];
  #pragma unroll
  for (int w = 1; w < 8; ++w) mx = fmaxf(mx, red[w]);
  __syncthreads();

  float sum = 0.f;
  for (int i = t; i < NKV; i += 256) {
    float v = (float)x[i]; v = v >= 0.f ? v : 0.01f * v;
    sum += __expf(v - mx);
  }
  for (int off = 16; off; off >>= 1) sum += __shfl_xor(sum, off, 32);
  if (ln == 0) red[wv] = sum;
  __syncthreads();
  sum = 0.f;
  #pragma unroll
  for (int w = 0; w < 8; ++w) sum += red[w];
  const float lse = mx + __logf(sum);

  if (mode == 0) {
    for (int i = t; i < NKV; i += 256) {
      float v = (float)x[i]; v = v >= 0.f ? v : 0.01f * v;
      Yflat[row * NKV + i] = (bf16)(v - lse);
    }
  } else {
    const long b = row >> 11, s = row & (SS - 1);
    for (int i = t; i < NKV; i += 256) {
      float v = (float)x[i]; v = v >= 0.f ? v : 0.01f * v;
      int h = i / DKV, e = i - h * DKV;
      Ypad[((b * HH + h) * SS + s) * (long)DKVP + e] = (bf16)(v - lse);
    }
    if (t < HH * 16) {
      int h = t >> 4, e = DKV + (t & 15);
      Ypad[((b * HH + h) * SS + s) * (long)DKVP + e] = (bf16)0.f;
    }
  }
}

// ---------------- generic batched GEMM: C = A(MxK) * B(NxK)^T + bias ----------------
// Block tile 128x128, 8 waves, wave tile 32x64 (2x4 WMMA tiles).
// Double-buffered LDS filled by global_load_async_to_lds_b128 (ASYNCcnt).
__global__ __launch_bounds__(256) void gemm_bf16_wmma(
    const bf16* __restrict__ A, const bf16* __restrict__ Bm,
    const float* __restrict__ bias, void* __restrict__ C,
    int K, int lda, int ldb, int ldc,
    long aBatch, long bBatch, long cBatch, int storeF32) {
  __shared__ bf16 lA[2][128][40];  // +16B row pad -> conflict-free frag reads
  __shared__ bf16 lB[2][128][40];
  const int m0 = blockIdx.x * 128;
  const int n0 = blockIdx.y * 128;
  const bf16* Ab = A + (long)blockIdx.z * aBatch;
  const bf16* Bb = Bm + (long)blockIdx.z * bBatch;
  const int t = threadIdx.x;
  const int lrow = t >> 1;            // 0..127
  const int lcg  = (t & 1) * 8;       // elem col 0 or 8 (second 16B via offset:32)
  const int wave = t >> 5, lane = t & 31;
  const int mrow = (wave >> 1) * 32;  // 0,32,64,96
  const int ncol = (wave & 1) * 64;   // 0,64
  const int fr = lane & 15;
  const int kb = (lane >> 4) * 8;

  f32x8 acc[2][4];
  #pragma unroll
  for (int i = 0; i < 2; ++i)
    #pragma unroll
    for (int jn = 0; jn < 4; ++jn) acc[i][jn] = (f32x8){};

  auto issue = [&](int kk, int p) {
    const bf16* ga = Ab + (long)(m0 + lrow) * lda + kk + lcg;
    const bf16* gb = Bb + (long)(n0 + lrow) * ldb + kk + lcg;
    async_cp32((unsigned)(unsigned long long)&lA[p][lrow][lcg], (const void*)ga);
    async_cp32((unsigned)(unsigned long long)&lB[p][lrow][lcg], (const void*)gb);
  };
  auto compute = [&](int p) {
    bf16x16 af[2], bf_[4];
    #pragma unroll
    for (int i = 0; i < 2; ++i)
      af[i] = CAT16(*(const bf16x8*)&lA[p][mrow + 16 * i + fr][kb],
                    *(const bf16x8*)&lA[p][mrow + 16 * i + fr][kb + 16]);
    #pragma unroll
    for (int jn = 0; jn < 4; ++jn)
      bf_[jn] = CAT16(*(const bf16x8*)&lB[p][ncol + 16 * jn + fr][kb],
                      *(const bf16x8*)&lB[p][ncol + 16 * jn + fr][kb + 16]);
    #pragma unroll
    for (int i = 0; i < 2; ++i)
      #pragma unroll
      for (int jn = 0; jn < 4; ++jn) acc[i][jn] = wmma_bf16(af[i], bf_[jn], acc[i][jn]);
  };

  int p = 0;
  issue(0, 0);
  wait_async0();
  __syncthreads();
  for (int kk = 32; kk < K; kk += 32) {
    issue(kk, p ^ 1);   // overlap next-tile async copy with compute
    compute(p);
    wait_async0();
    __syncthreads();
    p ^= 1;
  }
  compute(p);

  const int rb = m0 + mrow + (lane >> 4) * 8;
  #pragma unroll
  for (int jn = 0; jn < 4; ++jn) {
    const int col = n0 + ncol + 16 * jn + fr;
    const float bb = bias ? bias[col] : 0.f;
    if (storeF32) {
      float* Cp = (float*)C + (long)blockIdx.z * cBatch;
      #pragma unroll
      for (int i = 0; i < 2; ++i)
        #pragma unroll
        for (int j = 0; j < 8; ++j)
          Cp[(long)(rb + 16 * i + j) * ldc + col] = acc[i][jn][j] + bb;
    } else {
      bf16* Cp = (bf16*)C + (long)blockIdx.z * cBatch;
      #pragma unroll
      for (int i = 0; i < 2; ++i)
        #pragma unroll
        for (int j = 0; j < 8; ++j)
          Cp[(long)(rb + 16 * i + j) * ldc + col] = (bf16)(acc[i][jn][j] + bb);
    }
  }
}

// ---------------- fused scores + softmax per (b,h) ----------------
__global__ __launch_bounds__(256) void scores_softmax(
    const bf16* __restrict__ Q, const bf16* __restrict__ Kl, bf16* __restrict__ smpad) {
  __shared__ bf16 lQ[128][72];  // [d][s] transposed chunk, 144B row stride
  __shared__ bf16 lK[144][72];  // [e][s]
  const int bh = blockIdx.x;
  const long b = bh >> 3, h = bh & 7;
  const bf16* Qb = Q + (b * SS) * (long)DM + h * DK;
  const bf16* Kb = Kl + (b * SS) * (long)NKV + h * DKV;
  const int t = threadIdx.x;
  const int wave = t >> 5, lane = t & 31;
  const int fr = lane & 15, kb = (lane >> 4) * 8;
  const int sL = t >> 2, qg = (t & 3) * 8;
  const int sL2 = t >> 1, eg2 = 128 + (t & 1) * 8;

  f32x8 acc[9];
  #pragma unroll
  for (int e = 0; e < 9; ++e) acc[e] = (f32x8){};

  for (int s0 = 0; s0 < SS; s0 += 64) {
    bf16x8 qv[4], kv[4], kx;
    #pragma unroll
    for (int i = 0; i < 4; ++i) {
      qv[i] = *(const bf16x8*)(Qb + (long)(s0 + sL) * DM + qg + 32 * i);
      kv[i] = *(const bf16x8*)(Kb + (long)(s0 + sL) * NKV + qg + 32 * i);
    }
    if (t < 128) kx = *(const bf16x8*)(Kb + (long)(s0 + sL2) * NKV + eg2);
    __syncthreads();
    #pragma unroll
    for (int i = 0; i < 4; ++i)
      #pragma unroll
      for (int u = 0; u < 8; ++u) {
        lQ[qg + 32 * i + u][sL] = qv[i][u];
        lK[qg + 32 * i + u][sL] = kv[i][u];
      }
    if (t < 128)
      #pragma unroll
      for (int u = 0; u < 8; ++u) lK[eg2 + u][sL2] = kx[u];
    __syncthreads();

    #pragma unroll
    for (int sub = 0; sub < 64; sub += 32) {
      bf16x16 af = CAT16(*(const bf16x8*)&lQ[wave * 16 + fr][sub + kb],
                         *(const bf16x8*)&lQ[wave * 16 + fr][sub + kb + 16]);
      #pragma unroll
      for (int e = 0; e < 9; ++e) {
        bf16x16 bf_ = CAT16(*(const bf16x8*)&lK[e * 16 + fr][sub + kb],
                            *(const bf16x8*)&lK[e * 16 + fr][sub + kb + 16]);
        acc[e] = wmma_bf16(af, bf_, acc[e]);
      }
    }
  }

  const float scale = 1.0f / 12.0f;  // 1/sqrt(144)
  #pragma unroll
  for (int j = 0; j < 8; ++j) {
    float ev[9];
    float mx = -3.0e38f;
    #pragma unroll
    for (int e = 0; e < 9; ++e) { ev[e] = acc[e][j] * scale; mx = fmaxf(mx, ev[e]); }
    #pragma unroll
    for (int off = 1; off < 16; off <<= 1) mx = fmaxf(mx, __shfl_xor(mx, off, 32));
    float s = 0.f;
    #pragma unroll
    for (int e = 0; e < 9; ++e) { ev[e] = __expf(ev[e] - mx); s += ev[e]; }
    #pragma unroll
    for (int off = 1; off < 16; off <<= 1) s += __shfl_xor(s, off, 32);
    const float is = 1.0f / s;
    const int d = wave * 16 + j + ((lane >> 4) << 3);
    bf16* row = smpad + ((long)bh * DK + d) * DKVP;
    #pragma unroll
    for (int e = 0; e < 9; ++e) row[e * 16 + fr] = (bf16)(ev[e] * is);
    row[DKV + fr] = (bf16)0.f;
  }
}

// ---------------- host orchestration ----------------

extern "C" void kernel_launch(void* const* d_in, const int* in_sizes, int n_in,
                              void* d_out, int out_size, void* d_ws, size_t ws_size,
                              hipStream_t stream) {
  (void)in_sizes; (void)n_in; (void)out_size; (void)ws_size;
  const float* XQ = (const float*)d_in[0];
  const float* XK = (const float*)d_in[1];
  const float* XV = (const float*)d_in[2];
  const float* YK = (const float*)d_in[3];
  const float* YV = (const float*)d_in[4];
  const float* Wq = (const float*)d_in[6];
  const float* bq = (const float*)d_in[7];
  const float* Wk = (const float*)d_in[8];
  const float* bk = (const float*)d_in[9];
  const float* Wv = (const float*)d_in[10];
  const float* bv = (const float*)d_in[11];
  const float* Wo = (const float*)d_in[12];
  const float* bo = (const float*)d_in[13];

  char* ws = (char*)d_ws;
  bf16* XQbf  = (bf16*)(ws + 0L);          // reused as attnH
  bf16* attnH = XQbf;
  bf16* XKYbf = (bf16*)(ws + 33554432L);   // reused as attnPacked
  bf16* attnP = XKYbf;
  bf16* XVYbf = (bf16*)(ws + 71303168L);
  bf16* Qbf   = (bf16*)(ws + 109051904L);
  bf16* Kbf   = (bf16*)(ws + 142606336L);  // raw -> log_softmax in place
  bf16* Vbf   = (bf16*)(ws + 180355072L);
  bf16* Vpad  = (bf16*)(ws + 218103808L);  // [B,H,S,160]
  bf16* smpad = (bf16*)(ws + 260046848L);  // [64,128,160]
  bf16* Wqbf  = (bf16*)(ws + 262668288L);
  bf16* Wkbf  = (bf16*)(ws + 264765440L);
  bf16* Wvbf  = (bf16*)(ws + 267419648L);
  bf16* Wobf  = (bf16*)(ws + 270073856L);

  cvt_f32_bf16<<<2048, 256, 0, stream>>>(XQ, XQbf, ROWS * (long)DM);
  cvt_f32_bf16<<<1024, 256, 0, stream>>>(Wq, Wqbf, (long)DM * DM);
  cvt_f32_bf16<<<1024, 256, 0, stream>>>(Wk, Wkbf, (long)NKV * NKV);
  cvt_f32_bf16<<<1024, 256, 0, stream>>>(Wv, Wvbf, (long)NKV * NKV);
  cvt_f32_bf16<<<1024, 256, 0, stream>>>(Wo, Wobf, (long)DM * DM);
  pack_concat_bf16<<<2048, 256, 0, stream>>>(XK, YK, XKYbf);
  pack_concat_bf16<<<2048, 256, 0, stream>>>(XV, YV, XVYbf);

  gemm_bf16_wmma<<<dim3(128, 8, 1), 256, 0, stream>>>(
      XQbf, Wqbf, bq, Qbf, DM, DM, DM, DM, 0, 0, 0, 0);
  gemm_bf16_wmma<<<dim3(128, 9, 1), 256, 0, stream>>>(
      XKYbf, Wkbf, bk, Kbf, NKV, NKV, NKV, NKV, 0, 0, 0, 0);
  gemm_bf16_wmma<<<dim3(128, 9, 1), 256, 0, stream>>>(
      XVYbf, Wvbf, bv, Vbf, NKV, NKV, NKV, NKV, 0, 0, 0, 0);

  act_logsoftmax<<<ROWS, 256, 0, stream>>>(Kbf, Kbf, nullptr, 0);
  act_logsoftmax<<<ROWS, 256, 0, stream>>>(Vbf, nullptr, Vpad, 1);

  scores_softmax<<<64, 256, 0, stream>>>(Qbf, Kbf, smpad);

  gemm_bf16_wmma<<<dim3(16, 1, 64), 256, 0, stream>>>(
      Vpad, smpad, nullptr, attnH, DKVP, DKVP, DKVP, DK,
      (long)SS * DKVP, (long)DK * DKVP, (long)SS * DK, 0);

  pack_attn<<<2048, 256, 0, stream>>>(attnH, attnP);
  gemm_bf16_wmma<<<dim3(128, 8, 1), 256, 0, stream>>>(
      attnP, Wobf, bo, d_out, DM, DM, DM, DM, 0, 0, 0, 1);
}
